// GVPGNN_77446850281849
// MI455X (gfx1250) — matrix operand order repaired and checked by reading, not compiled, not run on hardware
//
#include <hip/hip_runtime.h>
#include <math.h>
#include <stdint.h>

// ---------------------------------------------------------------------------
// CDNA5 (gfx1250) GVP-GNN forward pass.
// All dense matmuls run on v_wmma_f32_16x16x32_f16 (wave32 WMMA), weights
// pre-packed to f16 fragment order, activations staged in LDS, fp32 accum.
// Node/edge features are mirrored in f16 so the bandwidth-dominant per-edge
// gathers move half the bytes and run as 128-bit loads.
// ---------------------------------------------------------------------------

typedef __attribute__((ext_vector_type(16))) _Float16 v16h;
typedef __attribute__((ext_vector_type(8)))  _Float16 v8h;
typedef __attribute__((ext_vector_type(8)))  float    v8f;

#define DEV __device__ __forceinline__

DEV float sigmoidf_(float x) { return 1.f / (1.f + __expf(-x)); }

// A fragment (16x32 f16) from row-major f16 buffer (LDS or global).
// Layout used here (contiguous k, same as B packing): lane -> m = lane&15, khalf = lane>>4;
// halves 0..15 -> K = k0 + khalf*16 + j
DEV v16h load_frag_a(const _Float16* A, int lda, int k0, int lane) {
  const int m = lane & 15;
  const int khalf = (lane >> 4) & 1;
  const _Float16* p = A + (size_t)m * lda + k0 + khalf * 16;
  v16h a;
#pragma unroll
  for (int j = 0; j < 16; ++j) { a[j] = p[j]; }
  return a;
}

// B fragment (32x16 f16) from packed weights: per (ktile,ntile) tile,
// 32 lanes x 16 contiguous halves; lane -> (kgroup = lane>>4, col = lane&15),
// half j -> K = kt*32 + kgroup*16 + j.
DEV v16h load_frag_b(const _Float16* B, int ntiles, int kt, int nt, int lane) {
  const _Float16* p = B + ((((size_t)kt * ntiles + nt) * 32 + lane) * 16);
  return *(const v16h*)p;
}

// Generic block GEMM over (mtiles x ntiles) 16x16 output tiles, K = ksteps*32.
// Jobs distributed across the block's waves. Store is a fused per-element sink:
// store(m, n, value) with C layout lane -> col = lane&15, rows (lane>>4)*8 + r.
template <typename StoreFn>
__device__ void gemm_tiles(const _Float16* A, int lda, const _Float16* B,
                           int mtiles, int ntiles, int ksteps,
                           const float* bias, StoreFn&& store) {
  const int wave = threadIdx.x >> 5;
  const int nw   = blockDim.x >> 5;
  const int lane = threadIdx.x & 31;
  const int njobs = mtiles * ntiles;
  for (int job = wave; job < njobs; job += nw) {
    const int mt = job / ntiles, nt = job % ntiles;
    const float binit = bias ? bias[nt * 16 + (lane & 15)] : 0.f;
    v8f c;
#pragma unroll
    for (int r = 0; r < 8; ++r) c[r] = binit;
    const _Float16* Arow = A + (size_t)mt * 16 * lda;
    for (int kt = 0; kt < ksteps; ++kt) {
      v16h a = load_frag_a(Arow, lda, kt * 32, lane);
      v16h b = load_frag_b(B, ntiles, kt, nt, lane);
      c = __builtin_amdgcn_wmma_f32_16x16x32_f16(false, a, false, b,
                                                 (short)0, c, false, false);
    }
    const int n = lane & 15, mbase = (lane >> 4) * 8;
#pragma unroll
    for (int r = 0; r < 8; ++r) store(mt * 16 + mbase + r, nt * 16 + n, c[r]);
  }
}

// ---------------------------------------------------------------------------
// Weight packing: row-major fp32 [K][N] -> f16 fragment order, zero padded.
// ---------------------------------------------------------------------------
__global__ __launch_bounds__(256) void pack_b_kernel(const float* __restrict__ src,
                                                     _Float16* __restrict__ dst,
                                                     int K, int Nn, int Kp, int Np) {
  int idx = blockIdx.x * blockDim.x + threadIdx.x;
  int total = Kp * Np;
  if (idx >= total) return;
  int kp = idx / Np, np_ = idx % Np;
  float val = (kp < K && np_ < Nn) ? src[(size_t)kp * Nn + np_] : 0.f;
  int kt = kp >> 5, kin = kp & 31, nt = np_ >> 4, nin = np_ & 15;
  int lane = ((kin >> 4) << 4) | nin;
  int j = kin & 15;
  int bnt = Np >> 4;
  size_t o = (((size_t)kt * bnt + nt) * 32 + lane) * 16 + j;
  dst[o] = (_Float16)val;
}

// ---------------------------------------------------------------------------
// Node embedding + ln_v + W_v GVP (small dims, one wave per node).
// Writes fp32 state + f16 mirror.
// ---------------------------------------------------------------------------
__global__ __launch_bounds__(256) void node_init_kernel(
    const float* __restrict__ node_s, const float* __restrict__ node_v,
    const float* __restrict__ res_emb,
    const float* __restrict__ lnv_g, const float* __restrict__ lnv_b,
    const float* __restrict__ wh, const float* __restrict__ ws_w,
    const float* __restrict__ ws_b, const float* __restrict__ wsv_w,
    const float* __restrict__ wsv_b, const float* __restrict__ wv,
    float* __restrict__ s_out, float* __restrict__ v_out,
    _Float16* __restrict__ s16, _Float16* __restrict__ v16, int N) {
  __shared__ float stmp[8][256];
  const int wave = threadIdx.x >> 5, lane = threadIdx.x & 31;
  const int node = blockIdx.x * 8 + wave;
  if (node >= N) return;
  // s = node_s @ res_emb (6x6), then LN over 6 dims (redundant per lane).
  float s0[6];
  for (int o = 0; o < 6; ++o) {
    float a = 0.f;
    for (int k = 0; k < 6; ++k) a += node_s[node * 6 + k] * res_emb[k * 6 + o];
    s0[o] = a;
  }
  float mu = 0.f;
  for (int o = 0; o < 6; ++o) mu += s0[o];
  mu /= 6.f;
  float var = 0.f;
  for (int o = 0; o < 6; ++o) { float d = s0[o] - mu; var += d * d; }
  var /= 6.f;
  float istd = rsqrtf(var + 1e-5f);
  for (int o = 0; o < 6; ++o) s0[o] = (s0[o] - mu) * istd * lnv_g[o] + lnv_b[o];
  // vector LN (3 channels x 3 coords)
  float vl[3][3]; float mnv = 0.f;
  for (int ch = 0; ch < 3; ++ch) {
    float nv = 0.f;
    for (int c = 0; c < 3; ++c) {
      float t = node_v[((size_t)node * 3 + ch) * 3 + c];
      vl[ch][c] = t; nv += t * t;
    }
    mnv += fmaxf(nv, 1e-8f);
  }
  mnv /= 3.f;
  float invn = rsqrtf(mnv);
  for (int ch = 0; ch < 3; ++ch)
    for (int c = 0; c < 3; ++c) vl[ch][c] *= invn;
  // vh (h = lane, 32 dims), vn
  float vh[3];
  for (int c = 0; c < 3; ++c)
    vh[c] = vl[0][c] * wh[lane] + vl[1][c] * wh[32 + lane] + vl[2][c] * wh[64 + lane];
  float vn = sqrtf(fmaxf(vh[0]*vh[0] + vh[1]*vh[1] + vh[2]*vh[2], 1e-8f));
  // s_out = concat(s0, vn) @ ws_w(38x256) + b ; 8 outputs per lane
  float acc[8];
  for (int j = 0; j < 8; ++j) {
    int o = j * 32 + lane;
    float a = ws_b[o];
    for (int k = 0; k < 6; ++k) a += s0[k] * ws_w[k * 256 + o];
    acc[j] = a;
  }
  for (int h = 0; h < 32; ++h) {
    float vnh = __shfl(vn, h);
    for (int j = 0; j < 8; ++j) {
      int o = j * 32 + lane;
      acc[j] += vnh * ws_w[(6 + h) * 256 + o];
    }
  }
  for (int j = 0; j < 8; ++j) {
    int o = j * 32 + lane;
    stmp[wave][o] = acc[j];
    s_out[(size_t)node * 256 + o] = acc[j];           // relu_s = False for W_v
    s16[(size_t)node * 256 + o] = (_Float16)acc[j];
  }
  // gate (vo = lane), K = 256
  float ga = wsv_b[lane];
  for (int c2 = 0; c2 < 256; ++c2) ga += stmp[wave][c2] * wsv_w[c2 * 32 + lane];
  ga = sigmoidf_(ga);
  // v_out (vo = lane) = (sum_h vh_h * wv[h][lane]) * gate
  float vo[3] = {0.f, 0.f, 0.f};
  for (int h = 0; h < 32; ++h) {
    float a0 = __shfl(vh[0], h), a1 = __shfl(vh[1], h), a2 = __shfl(vh[2], h);
    float w = wv[h * 32 + lane];
    vo[0] += a0 * w; vo[1] += a1 * w; vo[2] += a2 * w;
  }
  for (int c = 0; c < 3; ++c) {
    float val = vo[c] * ga;
    v_out[((size_t)node * 32 + lane) * 3 + c] = val;
    v16[((size_t)node * 32 + lane) * 3 + c] = (_Float16)val;
  }
}

// ---------------------------------------------------------------------------
// Edge features: vec, RBF, ln_e, W_e GVP, degree. One thread per edge.
// es/ev are only consumed as f16 -> store f16 directly.
// ---------------------------------------------------------------------------
__global__ __launch_bounds__(256) void edge_init_kernel(
    const float* __restrict__ pos, const int* __restrict__ src,
    const int* __restrict__ dst,
    const float* __restrict__ lne_g, const float* __restrict__ lne_b,
    const float* __restrict__ we_wh, const float* __restrict__ we_ws_w,
    const float* __restrict__ we_ws_b, const float* __restrict__ we_wsv_w,
    const float* __restrict__ we_wsv_b, const float* __restrict__ we_wv,
    _Float16* __restrict__ es_out, _Float16* __restrict__ ev_out,
    float* __restrict__ deg, int E) {
  int e = blockIdx.x * blockDim.x + threadIdx.x;
  if (e >= E) return;
  int si = src[e], di = dst[e];
  float vx = pos[di * 3 + 0] - pos[si * 3 + 0];
  float vy = pos[di * 3 + 1] - pos[si * 3 + 1];
  float vz = pos[di * 3 + 2] - pos[si * 3 + 2];
  float n2 = fmaxf(vx*vx + vy*vy + vz*vz, 1e-8f);
  float d = sqrtf(n2);
  // RBF(32) + LN over 32
  float rbf[32]; const float sigma_inv = 32.f / 20.f;
  float mu = 0.f;
  for (int k = 0; k < 32; ++k) {
    float m = (20.f / 31.f) * (float)k;
    float t = (d - m) * sigma_inv;
    rbf[k] = __expf(-t * t);
    mu += rbf[k];
  }
  mu /= 32.f;
  float var = 0.f;
  for (int k = 0; k < 32; ++k) { float dd = rbf[k] - mu; var += dd * dd; }
  var /= 32.f;
  float istd = rsqrtf(var + 1e-5f);
  for (int k = 0; k < 32; ++k) rbf[k] = (rbf[k] - mu) * istd * lne_g[k] + lne_b[k];
  // vector LN on ev (single channel -> unit vector)
  float inv = rsqrtf(n2);
  float ex = vx * inv, ey = vy * inv, ez = vz * inv;
  // W_e GVP: h = 1
  float w00 = we_wh[0];
  float hx = ex * w00, hy = ey * w00, hz = ez * w00;
  float vnh = sqrtf(fmaxf(hx*hx + hy*hy + hz*hz, 1e-8f));
  float so[32];
  float gdot = we_wsv_b[0];
  for (int o = 0; o < 32; ++o) {
    float a = we_ws_b[o];
    for (int k = 0; k < 32; ++k) a += rbf[k] * we_ws_w[k * 32 + o];
    a += vnh * we_ws_w[32 * 32 + o];
    so[o] = a;
    gdot += a * we_wsv_w[o];
  }
  float g = sigmoidf_(gdot);
  float wv0 = we_wv[0];
  for (int o = 0; o < 32; ++o)
    es_out[(size_t)e * 32 + o] = (_Float16)so[o];     // relu_s = False
  ev_out[(size_t)e * 3 + 0] = (_Float16)(hx * wv0 * g);
  ev_out[(size_t)e * 3 + 1] = (_Float16)(hy * wv0 * g);
  ev_out[(size_t)e * 3 + 2] = (_Float16)(hz * wv0 * g);
  atomicAdd(&deg[di], 1.f);
}

// ---------------------------------------------------------------------------
// LayerNorm (optionally adding ds/deg, dv/deg first). One wave per node.
// Refreshes the f16 mirrors.
// ---------------------------------------------------------------------------
__global__ __launch_bounds__(256) void ln_kernel(
    float* __restrict__ s, float* __restrict__ v,
    const float* __restrict__ dsb, const float* __restrict__ dvb,
    const float* __restrict__ deg,
    const float* __restrict__ gg, const float* __restrict__ bb,
    _Float16* __restrict__ s16, _Float16* __restrict__ v16, int N) {
  const int wave = threadIdx.x >> 5, lane = threadIdx.x & 31;
  const int node = blockIdx.x * 8 + wave;
  if (node >= N) return;
  float inv = 1.f;
  if (deg) inv = 1.f / fmaxf(deg[node], 1.f);
  const size_t sb = (size_t)node * 256;
  float x[8]; float sum = 0.f;
  for (int j = 0; j < 8; ++j) {
    int c = j * 32 + lane;
    float xv = s[sb + c];
    if (dsb) xv += dsb[sb + c] * inv;
    x[j] = xv; sum += xv;
  }
  for (int o = 16; o > 0; o >>= 1) sum += __shfl_xor(sum, o);
  float mu = sum / 256.f;
  float sum2 = 0.f;
  for (int j = 0; j < 8; ++j) { float d = x[j] - mu; sum2 += d * d; }
  for (int o = 16; o > 0; o >>= 1) sum2 += __shfl_xor(sum2, o);
  float istd = rsqrtf(sum2 / 256.f + 1e-5f);
  for (int j = 0; j < 8; ++j) {
    int c = j * 32 + lane;
    float val = (x[j] - mu) * istd * gg[c] + bb[c];
    s[sb + c] = val;
    s16[sb + c] = (_Float16)val;
  }
  // vectors: channel = lane (32 channels x 3 coords)
  const size_t vb = (size_t)node * 96;
  float vv[3]; float nv = 0.f;
  for (int c = 0; c < 3; ++c) {
    float t = v[vb + lane * 3 + c];
    if (dvb) t += dvb[vb + lane * 3 + c] * inv;
    vv[c] = t; nv += t * t;
  }
  nv = fmaxf(nv, 1e-8f);
  float m = nv;
  for (int o = 16; o > 0; o >>= 1) m += __shfl_xor(m, o);
  m /= 32.f;
  float invn = rsqrtf(m);
  for (int c = 0; c < 3; ++c) {
    float val = vv[c] * invn;
    v[vb + lane * 3 + c] = val;
    v16[vb + lane * 3 + c] = (_Float16)val;
  }
}

// ---------------------------------------------------------------------------
// Per-edge message GVP chain (3 GVPs) on WMMA; 16 edges / block, 8 waves.
// Gathers from f16 mirrors with 128-bit loads.
// ---------------------------------------------------------------------------
__global__ __launch_bounds__(256) void msg_kernel(
    const _Float16* __restrict__ s16, const _Float16* __restrict__ v16,
    const _Float16* __restrict__ es, const _Float16* __restrict__ ev,
    const int* __restrict__ src, const int* __restrict__ dst,
    const _Float16* wh0, const _Float16* ws0, const _Float16* wv0, const _Float16* wsv0,
    const _Float16* wh1, const _Float16* ws1, const _Float16* wv1, const _Float16* wsv1,
    const _Float16* wh2, const _Float16* ws2, const _Float16* wv2,
    const float* __restrict__ b_ws0, const float* __restrict__ b_wsv0,
    const float* __restrict__ b_ws1, const float* __restrict__ b_wsv1,
    const float* __restrict__ b_ws2,
    float* __restrict__ ds_acc, float* __restrict__ dv_acc) {
  __shared__ __align__(16) _Float16 Asc[16][640];  // s activations (K pad 640)
  __shared__ __align__(16) _Float16 Av[48][96];    // v input, rows = coord*16+edge
  __shared__ __align__(16) _Float16 Avh[48][96];   // vh
  __shared__ float Sout[16][256];
  __shared__ float Vout[48][32];
  __shared__ float Gate[16][32];

  const int t = threadIdx.x;
  const int e0 = blockIdx.x * 16;

  // ---- gather (vectorized 8-half chunks for the s rows) -------------------
  for (int i = t; i < 16 * 32; i += 256) {          // 16 edges x 32 chunks
    int e = i >> 5, c8 = (i & 31) * 8, ge = e0 + e;
    *(v8h*)&Asc[e][c8]       = *(const v8h*)&s16[(size_t)src[ge] * 256 + c8];
    *(v8h*)&Asc[e][288 + c8] = *(const v8h*)&s16[(size_t)dst[ge] * 256 + c8];
  }
  for (int i = t; i < 16 * 4; i += 256) {           // es: 4 chunks per edge
    int e = i >> 2, c8 = (i & 3) * 8;
    *(v8h*)&Asc[e][256 + c8] = *(const v8h*)&es[(size_t)(e0 + e) * 32 + c8];
  }
  for (int i = t; i < 16 * 31; i += 256) {
    int e = i / 31, c = i % 31;
    Asc[e][609 + c] = (_Float16)0.f;
  }
  for (int i = t; i < 48 * 32; i += 256) {
    int r = i >> 5, k = i & 31;
    int cc = r >> 4, e = r & 15, ge = e0 + e;
    Av[r][k]      = v16[((size_t)src[ge] * 32 + k) * 3 + cc];
    Av[r][33 + k] = v16[((size_t)dst[ge] * 32 + k) * 3 + cc];
  }
  for (int i = t; i < 48; i += 256) {
    int cc = i >> 4, e = i & 15;
    Av[i][32] = ev[(size_t)(e0 + e) * 3 + cc];
  }
  for (int i = t; i < 48 * 31; i += 256) {
    Av[i / 31][65 + i % 31] = (_Float16)0.f;
  }
  for (int i = t; i < 48 * 16; i += 256) {
    Avh[i >> 4][80 + (i & 15)] = (_Float16)0.f;
  }
  __syncthreads();

  // ---- GVP 0 (si=544, vi=65, h=65, so=256, vo=32, gate, relu) ------------
  gemm_tiles(&Av[0][0], 96, wh0, 3, 5, 3, nullptr,
             [&](int m, int n, float val) { Avh[m][n] = (_Float16)val; });
  __syncthreads();
  for (int i = t; i < 16 * 65; i += 256) {
    int e = i / 65, h = i % 65;
    float a0 = (float)Avh[e][h], a1 = (float)Avh[16 + e][h], a2 = (float)Avh[32 + e][h];
    Asc[e][544 + h] = (_Float16)sqrtf(fmaxf(a0*a0 + a1*a1 + a2*a2, 1e-8f));
  }
  __syncthreads();
  gemm_tiles(&Asc[0][0], 640, ws0, 1, 16, 20, b_ws0,
             [&](int m, int n, float val) { Sout[m][n] = val; });
  __syncthreads();
  for (int i = t; i < 16 * 256; i += 256) {
    int e = i >> 8, c = i & 255;
    Asc[e][c] = (_Float16)Sout[e][c];       // pre-relu f16 (gate input)
  }
  __syncthreads();
  gemm_tiles(&Asc[0][0], 640, wsv0, 1, 2, 8, b_wsv0,
             [&](int m, int n, float val) { Gate[m][n] = val; });
  gemm_tiles(&Avh[0][0], 96, wv0, 3, 2, 3, nullptr,
             [&](int m, int n, float val) { Vout[m][n] = val; });
  __syncthreads();
  for (int i = t; i < 48 * 32; i += 256) {
    int r = i >> 5, k = i & 31, e = r & 15;
    Av[r][k] = (_Float16)(Vout[r][k] * sigmoidf_(Gate[e][k]));
  }
  for (int i = t; i < 16 * 256; i += 256) {
    int e = i >> 8, c = i & 255;
    Asc[e][c] = (_Float16)fmaxf((float)Asc[e][c], 0.f);   // relu
  }
  __syncthreads();

  // ---- GVP 1 (si=256, vi=32, h=32, so=256, vo=32, gate, relu) ------------
  gemm_tiles(&Av[0][0], 96, wh1, 3, 2, 1, nullptr,
             [&](int m, int n, float val) { Avh[m][n] = (_Float16)val; });
  __syncthreads();
  for (int i = t; i < 16 * 32; i += 256) {
    int e = i >> 5, h = i & 31;
    float a0 = (float)Avh[e][h], a1 = (float)Avh[16 + e][h], a2 = (float)Avh[32 + e][h];
    Asc[e][256 + h] = (_Float16)sqrtf(fmaxf(a0*a0 + a1*a1 + a2*a2, 1e-8f));
  }
  __syncthreads();
  gemm_tiles(&Asc[0][0], 640, ws1, 1, 16, 9, b_ws1,
             [&](int m, int n, float val) { Sout[m][n] = val; });
  __syncthreads();
  for (int i = t; i < 16 * 256; i += 256) {
    int e = i >> 8, c = i & 255;
    Asc[e][c] = (_Float16)Sout[e][c];
  }
  __syncthreads();
  gemm_tiles(&Asc[0][0], 640, wsv1, 1, 2, 8, b_wsv1,
             [&](int m, int n, float val) { Gate[m][n] = val; });
  gemm_tiles(&Avh[0][0], 96, wv1, 3, 2, 1, nullptr,
             [&](int m, int n, float val) { Vout[m][n] = val; });
  __syncthreads();
  for (int i = t; i < 48 * 32; i += 256) {
    int r = i >> 5, k = i & 31, e = r & 15;
    Av[r][k] = (_Float16)(Vout[r][k] * sigmoidf_(Gate[e][k]));
  }
  for (int i = t; i < 16 * 256; i += 256) {
    int e = i >> 8, c = i & 255;
    Asc[e][c] = (_Float16)fmaxf((float)Asc[e][c], 0.f);
  }
  __syncthreads();

  // ---- GVP 2 (no gate, no relu) ------------------------------------------
  gemm_tiles(&Av[0][0], 96, wh2, 3, 2, 1, nullptr,
             [&](int m, int n, float val) { Avh[m][n] = (_Float16)val; });
  __syncthreads();
  for (int i = t; i < 16 * 32; i += 256) {
    int e = i >> 5, h = i & 31;
    float a0 = (float)Avh[e][h], a1 = (float)Avh[16 + e][h], a2 = (float)Avh[32 + e][h];
    Asc[e][256 + h] = (_Float16)sqrtf(fmaxf(a0*a0 + a1*a1 + a2*a2, 1e-8f));
  }
  __syncthreads();
  gemm_tiles(&Asc[0][0], 640, ws2, 1, 16, 9, b_ws2,
             [&](int m, int n, float val) { Sout[m][n] = val; });
  gemm_tiles(&Avh[0][0], 96, wv2, 3, 2, 1, nullptr,
             [&](int m, int n, float val) { Vout[m][n] = val; });
  __syncthreads();

  // ---- scatter-accumulate (segment sum over dst) --------------------------
  for (int i = t; i < 16 * 256; i += 256) {
    int e = i >> 8, c = i & 255, d = dst[e0 + e];
    atomicAdd(&ds_acc[(size_t)d * 256 + c], Sout[e][c]);
  }
  for (int i = t; i < 48 * 32; i += 256) {
    int r = i >> 5, k = i & 31, cc = r >> 4, e = r & 15;
    int d = dst[e0 + e];
    atomicAdd(&dv_acc[((size_t)d * 32 + k) * 3 + cc], Vout[r][k]);
  }
}

// ---------------------------------------------------------------------------
// FF GVP 0: (256,32) -> (1024,64), gate+relu. 16 nodes / block.
// ---------------------------------------------------------------------------
__global__ __launch_bounds__(256) void ff0_kernel(
    const _Float16* __restrict__ s16, const _Float16* __restrict__ v16,
    const _Float16* wh, const _Float16* ws, const _Float16* wv, const _Float16* wsv,
    const float* __restrict__ b_ws, const float* __restrict__ b_wsv,
    _Float16* __restrict__ s1g, _Float16* __restrict__ v1g) {
  __shared__ __align__(16) _Float16 Ain[16][320];
  __shared__ __align__(16) _Float16 Av[48][32];
  __shared__ __align__(16) _Float16 Avh[48][64];
  __shared__ __align__(16) _Float16 S1[16][1024];
  __shared__ float Gate[16][64];
  const int t = threadIdx.x;
  const int n0 = blockIdx.x * 16;
  for (int i = t; i < 16 * 32; i += 256) {
    int e = i >> 5, c8 = (i & 31) * 8;
    *(v8h*)&Ain[e][c8] = *(const v8h*)&s16[(size_t)(n0 + e) * 256 + c8];
  }
  for (int i = t; i < 48 * 32; i += 256) {
    int r = i >> 5, k = i & 31, cc = r >> 4, e = r & 15;
    Av[r][k] = v16[((size_t)(n0 + e) * 32 + k) * 3 + cc];
  }
  __syncthreads();
  gemm_tiles(&Av[0][0], 32, wh, 3, 4, 1, nullptr,
             [&](int m, int n, float val) { Avh[m][n] = (_Float16)val; });
  __syncthreads();
  for (int i = t; i < 16 * 64; i += 256) {
    int e = i >> 6, h = i & 63;
    float a0 = (float)Avh[e][h], a1 = (float)Avh[16 + e][h], a2 = (float)Avh[32 + e][h];
    Ain[e][256 + h] = (_Float16)sqrtf(fmaxf(a0*a0 + a1*a1 + a2*a2, 1e-8f));
  }
  __syncthreads();
  gemm_tiles(&Ain[0][0], 320, ws, 1, 64, 10, b_ws,
             [&](int m, int n, float val) { S1[m][n] = (_Float16)val; });
  __syncthreads();
  gemm_tiles(&S1[0][0], 1024, wsv, 1, 4, 32, b_wsv,
             [&](int m, int n, float val) { Gate[m][n] = val; });
  __syncthreads();
  gemm_tiles(&Avh[0][0], 64, wv, 3, 4, 2, nullptr,
             [&](int m, int n, float val) {
               int e = m & 15, cc = m >> 4;
               float g = sigmoidf_(Gate[e][n]);
               v1g[(((size_t)(n0 + e)) * 64 + n) * 3 + cc] = (_Float16)(val * g);
             });
  for (int i = t; i < 16 * 1024; i += 256) {
    int e = i >> 10, c = i & 1023;
    s1g[(size_t)(n0 + e) * 1024 + c] = (_Float16)fmaxf((float)S1[e][c], 0.f);
  }
}

// ---------------------------------------------------------------------------
// FF GVP 1: (1024,64) -> (256,32), gate, no relu. 16 nodes / block.
// ---------------------------------------------------------------------------
__global__ __launch_bounds__(256) void ff1_kernel(
    const _Float16* __restrict__ s1g, const _Float16* __restrict__ v1g,
    const _Float16* wh, const _Float16* ws, const _Float16* wv, const _Float16* wsv,
    const float* __restrict__ b_ws, const float* __restrict__ b_wsv,
    float* __restrict__ fs, float* __restrict__ fv) {
  __shared__ __align__(16) _Float16 Asc[16][1088];
  __shared__ __align__(16) _Float16 Av[48][64];
  __shared__ __align__(16) _Float16 Avh[48][64];
  __shared__ __align__(16) _Float16 SG[16][256];
  __shared__ float Gate[16][32];
  const int t = threadIdx.x;
  const int n0 = blockIdx.x * 16;
  for (int i = t; i < 16 * 128; i += 256) {
    int e = i >> 7, c8 = (i & 127) * 8;
    *(v8h*)&Asc[e][c8] = *(const v8h*)&s1g[(size_t)(n0 + e) * 1024 + c8];
  }
  for (int i = t; i < 48 * 64; i += 256) {
    int r = i >> 6, k = i & 63, cc = r >> 4, e = r & 15;
    Av[r][k] = v1g[(((size_t)(n0 + e)) * 64 + k) * 3 + cc];
  }
  __syncthreads();
  gemm_tiles(&Av[0][0], 64, wh, 3, 4, 2, nullptr,
             [&](int m, int n, float val) { Avh[m][n] = (_Float16)val; });
  __syncthreads();
  for (int i = t; i < 16 * 64; i += 256) {
    int e = i >> 6, h = i & 63;
    float a0 = (float)Avh[e][h], a1 = (float)Avh[16 + e][h], a2 = (float)Avh[32 + e][h];
    Asc[e][1024 + h] = (_Float16)sqrtf(fmaxf(a0*a0 + a1*a1 + a2*a2, 1e-8f));
  }
  __syncthreads();
  gemm_tiles(&Asc[0][0], 1088, ws, 1, 16, 34, b_ws,
             [&](int m, int n, float val) {
               SG[m][n] = (_Float16)val;
               fs[(size_t)(n0 + m) * 256 + n] = val;   // relu_s = False
             });
  __syncthreads();
  gemm_tiles(&SG[0][0], 256, wsv, 1, 2, 8, b_wsv,
             [&](int m, int n, float val) { Gate[m][n] = val; });
  __syncthreads();
  gemm_tiles(&Avh[0][0], 64, wv, 3, 2, 2, nullptr,
             [&](int m, int n, float val) {
               int e = m & 15, cc = m >> 4;
               float g = sigmoidf_(Gate[e][n]);
               fv[(((size_t)(n0 + e)) * 32 + n) * 3 + cc] = val * g;
             });
}

// ---------------------------------------------------------------------------
// Output GVP (vo=0): node_feature + graph segment sum. 16 nodes / block.
// ---------------------------------------------------------------------------
__global__ __launch_bounds__(256) void out_kernel(
    const _Float16* __restrict__ s16, const _Float16* __restrict__ v16,
    const _Float16* pwh, const _Float16* pws, const float* __restrict__ b_ws,
    const int* __restrict__ n2g, float* __restrict__ out_graph,
    float* __restrict__ out_node) {
  __shared__ __align__(16) _Float16 Asc[16][288];
  __shared__ __align__(16) _Float16 Av[48][32];
  __shared__ __align__(16) _Float16 Avh[48][32];
  const int t = threadIdx.x;
  const int n0 = blockIdx.x * 16;
  for (int i = t; i < 16 * 32; i += 256) {
    int e = i >> 5, c8 = (i & 31) * 8;
    *(v8h*)&Asc[e][c8] = *(const v8h*)&s16[(size_t)(n0 + e) * 256 + c8];
  }
  for (int i = t; i < 48 * 32; i += 256) {
    int r = i >> 5, k = i & 31, cc = r >> 4, e = r & 15;
    Av[r][k] = v16[((size_t)(n0 + e) * 32 + k) * 3 + cc];
  }
  __syncthreads();
  gemm_tiles(&Av[0][0], 32, pwh, 3, 2, 1, nullptr,
             [&](int m, int n, float val) { Avh[m][n] = (_Float16)val; });
  __syncthreads();
  for (int i = t; i < 16 * 32; i += 256) {
    int e = i >> 5, h = i & 31;
    float a0 = (float)Avh[e][h], a1 = (float)Avh[16 + e][h], a2 = (float)Avh[32 + e][h];
    Asc[e][256 + h] = (_Float16)sqrtf(fmaxf(a0*a0 + a1*a1 + a2*a2, 1e-8f));
  }
  __syncthreads();
  gemm_tiles(&Asc[0][0], 288, pws, 1, 16, 9, b_ws,
             [&](int m, int n, float val) {
               float r = fmaxf(val, 0.f);            // relu_s = True
               int node = n0 + m;
               out_node[(size_t)node * 256 + n] = r;
               atomicAdd(&out_graph[(size_t)n2g[node] * 256 + n], r);
             });
}

// ---------------------------------------------------------------------------
// Host orchestration
// ---------------------------------------------------------------------------
extern "C" void kernel_launch(void* const* d_in, const int* in_sizes, int n_in,
                              void* d_out, int out_size, void* d_ws, size_t ws_size,
                              hipStream_t stream) {
  (void)in_sizes; (void)out_size; (void)ws_size;
  constexpr int N = 10000, E = 300000;

  int ii = 0;
  const float* node_s = (const float*)d_in[ii++];
  const float* node_v = (const float*)d_in[ii++];
  const float* pos    = (const float*)d_in[ii++];
  const int*   eidx   = (const int*)d_in[ii++];
  const int*   n2g    = (const int*)d_in[ii++];
  const int* src = eidx;
  const int* dst = eidx + E;

  const float* P[160];
  const int nP = n_in - 5;
  for (int i = 0; i < nP && i < 160; ++i) P[i] = (const float*)d_in[5 + i];
  int p = 0;
  // params leaves, jax pytree (sorted-key) order
  const float *We_wh = P[p++], *We_ws_b = P[p++], *We_ws_w = P[p++],
              *We_wsv_b = P[p++], *We_wsv_w = P[p++], *We_wv = P[p++];
  const float *Wo_wh = P[p++], *Wo_ws_b = P[p++], *Wo_ws_w = P[p++];
  const float *Wv_wh = P[p++], *Wv_ws_b = P[p++], *Wv_ws_w = P[p++],
              *Wv_wsv_b = P[p++], *Wv_wsv_w = P[p++], *Wv_wv = P[p++];
  struct LayerP {
    const float *ff0_wh, *ff0_ws_b, *ff0_ws_w, *ff0_wsv_b, *ff0_wsv_w, *ff0_wv;
    const float *ff1_wh, *ff1_ws_b, *ff1_ws_w, *ff1_wsv_b, *ff1_wsv_w, *ff1_wv;
    const float *ln0_b, *ln0_g, *ln1_b, *ln1_g;
    const float *m0_wh, *m0_ws_b, *m0_ws_w, *m0_wsv_b, *m0_wsv_w, *m0_wv;
    const float *m1_wh, *m1_ws_b, *m1_ws_w, *m1_wsv_b, *m1_wsv_w, *m1_wv;
    const float *m2_wh, *m2_ws_b, *m2_ws_w, *m2_wv;
  } L[3];
  for (int l = 0; l < 3; ++l) {
    L[l].ff0_wh = P[p++]; L[l].ff0_ws_b = P[p++]; L[l].ff0_ws_w = P[p++];
    L[l].ff0_wsv_b = P[p++]; L[l].ff0_wsv_w = P[p++]; L[l].ff0_wv = P[p++];
    L[l].ff1_wh = P[p++]; L[l].ff1_ws_b = P[p++]; L[l].ff1_ws_w = P[p++];
    L[l].ff1_wsv_b = P[p++]; L[l].ff1_wsv_w = P[p++]; L[l].ff1_wv = P[p++];
    L[l].ln0_b = P[p++]; L[l].ln0_g = P[p++]; L[l].ln1_b = P[p++]; L[l].ln1_g = P[p++];
    L[l].m0_wh = P[p++]; L[l].m0_ws_b = P[p++]; L[l].m0_ws_w = P[p++];
    L[l].m0_wsv_b = P[p++]; L[l].m0_wsv_w = P[p++]; L[l].m0_wv = P[p++];
    L[l].m1_wh = P[p++]; L[l].m1_ws_b = P[p++]; L[l].m1_ws_w = P[p++];
    L[l].m1_wsv_b = P[p++]; L[l].m1_wsv_w = P[p++]; L[l].m1_wv = P[p++];
    L[l].m2_wh = P[p++]; L[l].m2_ws_b = P[p++]; L[l].m2_ws_w = P[p++]; L[l].m2_wv = P[p++];
  }
  const float *lne_b = P[p++], *lne_g = P[p++];
  const float *lno_b = P[p++], *lno_g = P[p++];
  const float *lnv_b = P[p++], *lnv_g = P[p++];
  const float *res_emb = P[p++];

  // workspace bump allocator (256B aligned)
  uintptr_t wp = ((uintptr_t)d_ws + 255) & ~(uintptr_t)255;
  auto alloc = [&](size_t bytes) -> void* {
    void* r = (void*)wp; wp += (bytes + 255) & ~(size_t)255; return r;
  };
  float* sbuf  = (float*)alloc((size_t)N * 256 * 4);
  float* vbuf  = (float*)alloc((size_t)N * 96 * 4);
  float* deg   = (float*)alloc((size_t)N * 4);
  float* dsb   = (float*)alloc((size_t)N * 256 * 4);
  float* dvb   = (float*)alloc((size_t)N * 96 * 4);
  float* fsb   = (float*)alloc((size_t)N * 256 * 4);
  float* fvb   = (float*)alloc((size_t)N * 96 * 4);
  _Float16* s16b = (_Float16*)alloc((size_t)N * 256 * 2);
  _Float16* v16b = (_Float16*)alloc((size_t)N * 96 * 2);
  _Float16* esb  = (_Float16*)alloc((size_t)E * 32 * 2);
  _Float16* evb  = (_Float16*)alloc((size_t)E * 3 * 2);
  _Float16* s1g  = (_Float16*)alloc((size_t)N * 1024 * 2);
  _Float16* v1g  = (_Float16*)alloc((size_t)N * 192 * 2);

  auto pack = [&](const float* w, int K, int Nn, int Kp, int Np) -> const _Float16* {
    _Float16* dstp = (_Float16*)alloc((size_t)Kp * Np * 2);
    int tot = Kp * Np;
    pack_b_kernel<<<(tot + 255) / 256, 256, 0, stream>>>(w, dstp, K, Nn, Kp, Np);
    return dstp;
  };

  struct Pk {
    const _Float16 *wh0, *ws0, *wv0, *wsv0, *wh1, *ws1, *wv1, *wsv1, *wh2, *ws2, *wv2;
    const _Float16 *f0wh, *f0ws, *f0wv, *f0wsv, *f1wh, *f1ws, *f1wv, *f1wsv;
  } pk[3];
  for (int l = 0; l < 3; ++l) {
    pk[l].wh0  = pack(L[l].m0_wh,    65, 65,   96, 80);
    pk[l].ws0  = pack(L[l].m0_ws_w, 609, 256, 640, 256);
    pk[l].wv0  = pack(L[l].m0_wv,    65, 32,   96, 32);
    pk[l].wsv0 = pack(L[l].m0_wsv_w, 256, 32, 256, 32);
    pk[l].wh1  = pack(L[l].m1_wh,    32, 32,   32, 32);
    pk[l].ws1  = pack(L[l].m1_ws_w, 288, 256, 288, 256);
    pk[l].wv1  = pack(L[l].m1_wv,    32, 32,   32, 32);
    pk[l].wsv1 = pack(L[l].m1_wsv_w, 256, 32, 256, 32);
    pk[l].wh2  = pack(L[l].m2_wh,    32, 32,   32, 32);
    pk[l].ws2  = pack(L[l].m2_ws_w, 288, 256, 288, 256);
    pk[l].wv2  = pack(L[l].m2_wv,    32, 32,   32, 32);
    pk[l].f0wh  = pack(L[l].ff0_wh,    32, 64,    32, 64);
    pk[l].f0ws  = pack(L[l].ff0_ws_w, 320, 1024, 320, 1024);
    pk[l].f0wv  = pack(L[l].ff0_wv,    64, 64,    64, 64);
    pk[l].f0wsv = pack(L[l].ff0_wsv_w, 1024, 64, 1024, 64);
    pk[l].f1wh  = pack(L[l].ff1_wh,    64, 64,    64, 64);
    pk[l].f1ws  = pack(L[l].ff1_ws_w, 1088, 256, 1088, 256);
    pk[l].f1wv  = pack(L[l].ff1_wv,    64, 32,    64, 32);
    pk[l].f1wsv = pack(L[l].ff1_wsv_w, 256, 32,  256, 32);
  }
  const _Float16* out_wh = pack(Wo_wh, 32, 32, 32, 32);
  const _Float16* out_ws = pack(Wo_ws_w, 288, 256, 288, 256);

  hipMemsetAsync(deg, 0, (size_t)N * 4, stream);
  hipMemsetAsync(d_out, 0, 8 * 256 * 4, stream);   // graph-feature accumulator

  node_init_kernel<<<(N + 7) / 8, 256, 0, stream>>>(
      node_s, node_v, res_emb, lnv_g, lnv_b,
      Wv_wh, Wv_ws_w, Wv_ws_b, Wv_wsv_w, Wv_wsv_b, Wv_wv,
      sbuf, vbuf, s16b, v16b, N);
  edge_init_kernel<<<(E + 255) / 256, 256, 0, stream>>>(
      pos, src, dst, lne_g, lne_b,
      We_wh, We_ws_w, We_ws_b, We_wsv_w, We_wsv_b, We_wv, esb, evb, deg, E);

  for (int l = 0; l < 3; ++l) {
    hipMemsetAsync(dsb, 0, (size_t)N * 256 * 4, stream);
    hipMemsetAsync(dvb, 0, (size_t)N * 96 * 4, stream);
    msg_kernel<<<E / 16, 256, 0, stream>>>(
        s16b, v16b, esb, evb, src, dst,
        pk[l].wh0, pk[l].ws0, pk[l].wv0, pk[l].wsv0,
        pk[l].wh1, pk[l].ws1, pk[l].wv1, pk[l].wsv1,
        pk[l].wh2, pk[l].ws2, pk[l].wv2,
        L[l].m0_ws_b, L[l].m0_wsv_b, L[l].m1_ws_b, L[l].m1_wsv_b, L[l].m2_ws_b,
        dsb, dvb);
    ln_kernel<<<(N + 7) / 8, 256, 0, stream>>>(sbuf, vbuf, dsb, dvb, deg,
                                               L[l].ln0_g, L[l].ln0_b,
                                               s16b, v16b, N);
    ff0_kernel<<<N / 16, 256, 0, stream>>>(
        s16b, v16b, pk[l].f0wh, pk[l].f0ws, pk[l].f0wv, pk[l].f0wsv,
        L[l].ff0_ws_b, L[l].ff0_wsv_b, s1g, v1g);
    ff1_kernel<<<N / 16, 256, 0, stream>>>(
        s1g, v1g, pk[l].f1wh, pk[l].f1ws, pk[l].f1wv, pk[l].f1wsv,
        L[l].ff1_ws_b, L[l].ff1_wsv_b, fsb, fvb);
    ln_kernel<<<(N + 7) / 8, 256, 0, stream>>>(sbuf, vbuf, fsb, fvb, nullptr,
                                               L[l].ln1_g, L[l].ln1_b,
                                               s16b, v16b, N);
  }

  ln_kernel<<<(N + 7) / 8, 256, 0, stream>>>(sbuf, vbuf, nullptr, nullptr, nullptr,
                                             lno_g, lno_b, s16b, v16b, N);
  float* out_graph = (float*)d_out;
  float* out_node  = (float*)d_out + 8 * 256;
  out_kernel<<<N / 16, 256, 0, stream>>>(s16b, v16b, out_wh, out_ws, Wo_ws_b,
                                         n2g, out_graph, out_node);
}
